// Similarity_Loss_1_vs_all_4183298146521
// MI455X (gfx1250) — matrix operand order repaired and checked by reading, not compiled, not run on hardware
//
#include <hip/hip_runtime.h>
#include <hip/hip_bf16.h>
#include <stdint.h>

// Problem constants (match reference setup_inputs)
#define B_SZ 8192
#define D_SZ 256
#define INV_TEMP 20.0f   // 1 / 0.05

typedef __attribute__((ext_vector_type(16))) __bf16 v16bf;
typedef __attribute__((ext_vector_type(8)))  __bf16 v8bf;
typedef __attribute__((ext_vector_type(8)))  float  v8f;
typedef __attribute__((ext_vector_type(4)))  int    v4i;

// ---------------------------------------------------------------------------
// CDNA5 async global->LDS path (ASYNCcnt), with a safe synchronous fallback.
// Probe result (round 2 diagnostic): param 0 of the async builtin is
// 'int4 addrspace(1)*' (printed as 'int __vector(4) __device__ *'), non-const.
// ---------------------------------------------------------------------------
#if defined(__AMDGCN__) && __has_builtin(__builtin_amdgcn_global_load_async_to_lds_b128)
#define HAS_ASYNC_LDS 1
#else
#define HAS_ASYNC_LDS 0
#endif

#if HAS_ASYNC_LDS
#if __has_builtin(__builtin_amdgcn_s_wait_asynccnt)
#define WAIT_ASYNC(n) __builtin_amdgcn_s_wait_asynccnt(n)
#else
#define WAIT_ASYNC(n) asm volatile("s_wait_asynccnt " #n ::: "memory")
#endif
#else
#define WAIT_ASYNC(n) ((void)0)
#endif

#define AS1_ __attribute__((address_space(1)))
#define AS3_ __attribute__((address_space(3)))

__device__ __forceinline__ void copy16B_to_lds(unsigned short* ldsDst,
                                               const unsigned short* gSrc) {
#if HAS_ASYNC_LDS
    __builtin_amdgcn_global_load_async_to_lds_b128(
        (AS1_ v4i*)gSrc,      // global source (addrspace(1) int4*, non-const)
        (AS3_ v4i*)ldsDst,    // LDS destination (addrspace(3) int4*)
        0, 0);
#else
    *(uint4*)ldsDst = *(const uint4*)gSrc;   // sync fallback (ds_store_b128)
#endif
}

__device__ __forceinline__ unsigned short f32_to_bf16_rne(float f) {
    uint32_t u = __float_as_uint(f);
    uint32_t r = u + 0x7FFFu + ((u >> 16) & 1u);   // round-to-nearest-even
    return (unsigned short)(r >> 16);
}

// ---------------------------------------------------------------------------
// Kernel 1: fp32 -> bf16 conversion + exact fp32 row squared-norms.
// One block per row pair (text row i, image row i), 256 threads = D.
// ---------------------------------------------------------------------------
__global__ void __launch_bounds__(256)
prep_kernel(const float* __restrict__ text, const float* __restrict__ img,
            unsigned short* __restrict__ textb, unsigned short* __restrict__ imgb,
            float* __restrict__ t2, float* __restrict__ i2) {
    const int row = blockIdx.x;
    const int tid = threadIdx.x;
    const size_t idx = (size_t)row * D_SZ + tid;

    float tv = text[idx];
    float iv = img[idx];
    textb[idx] = f32_to_bf16_rne(tv);
    imgb[idx]  = f32_to_bf16_rne(iv);

    __shared__ float st[256];
    __shared__ float si[256];
    st[tid] = tv * tv;
    si[tid] = iv * iv;
    __syncthreads();
    #pragma unroll
    for (int s = 128; s > 0; s >>= 1) {
        if (tid < s) { st[tid] += st[tid + s]; si[tid] += si[tid + s]; }
        __syncthreads();
    }
    if (tid == 0) { t2[row] = st[0]; i2[row] = si[0]; }
}

// ---------------------------------------------------------------------------
// Kernel 2: WMMA bf16 GEMM (gram = image @ text^T) with async-LDS-staged,
// double-buffered B panel, fused distance/exp epilogue, deterministic
// per-row partial sums and diagonal capture.
//
// Block: 256 threads = 8 wave32s stacked along M. Block tile = 128x128.
// Wave tile = 16(M) x 128(N) = 8 accumulators v8f; A fragment reused x8;
// B panel (128 cols x 32 K) shared by all 8 waves via LDS.
// ---------------------------------------------------------------------------
#define LDSB_STRIDE 40   // 32 K-elements + 8 pad: lane dword stride 20 -> 16
                         // distinct bank groups for lanes 0..15; 16B aligned.

__global__ void __launch_bounds__(256, 2)
gemm_sim_kernel(const unsigned short* __restrict__ imgb,   // A rows  = i
                const unsigned short* __restrict__ textb,  // B^T rows = j
                const float* __restrict__ i2, const float* __restrict__ t2,
                float* __restrict__ partial,               // [gridY][B_SZ]
                float* __restrict__ diag) {
    __shared__ unsigned short ldsB[2][128 * LDSB_STRIDE];  // 20 KB

    const int tid     = threadIdx.x;
    const int lane    = tid & 31;
    const int wave    = tid >> 5;              // 0..7  (M stacking)
    const int laneLo  = lane & 15;
    const int laneHi  = lane >> 4;             // 0 or 1

    const int rowBase = blockIdx.x * 128 + wave * 16;
    const int colBase = blockIdx.y * 128;

    // --- B panel staging map: 512 16-byte pieces, 2 per thread ------------
    // piece p in {tid, tid+256}: col = p>>2 (0..127), sub = p&3 (16B chunk).
    const int pc  = tid >> 2;                  // col for piece 0 (0..63)
    const int ps  = (tid & 3) * 8;             // element offset of 16B chunk

    // --- A fragment addressing (ISA 16-bit 16x32 A layout) ----------------
    //   lanes 0-15 : row = rowBase+laneLo, K = {k..k+7} U {k+16..k+23}
    //   lanes 16-31: same rows,            K = {k+8..k+15} U {k+24..k+31}
    const unsigned short* aRowPtr = imgb + (size_t)(rowBase + laneLo) * D_SZ;
    const int aKSub = laneHi * 8;

    v8f acc[8] = {};

    auto issue_panel = [&](int k, int bufSel) {
        const unsigned short* g0 = textb + (size_t)(colBase + pc) * D_SZ + k + ps;
        const unsigned short* g1 = textb + (size_t)(colBase + pc + 64) * D_SZ + k + ps;
        copy16B_to_lds(&ldsB[bufSel][pc * LDSB_STRIDE + ps], g0);
        copy16B_to_lds(&ldsB[bufSel][(pc + 64) * LDSB_STRIDE + ps], g1);
    };

    auto compute_step = [&](int k, int bufSel) {
        const int ka = k + aKSub;
        v8bf alo = *(const v8bf*)(aRowPtr + ka);
        v8bf ahi = *(const v8bf*)(aRowPtr + ka + 16);
        v16bf afrag = __builtin_shufflevector(alo, ahi,
            0,1,2,3,4,5,6,7,8,9,10,11,12,13,14,15);
        if (k + 32 < D_SZ)   // warm next A slice (global_prefetch_b8)
            __builtin_prefetch(aRowPtr + ka + 32, 0, 1);

        // B fragment (matches A): lanes 0-15: N=laneLo, K={k..k+7} U {k+16..k+23};
        // lanes 16-31: N=laneLo, K={k+8..k+15} U {k+24..k+31}.  From LDS.
        const unsigned short* bbase =
            &ldsB[bufSel][laneLo * LDSB_STRIDE + laneHi * 8];
        #pragma unroll
        for (int n = 0; n < 8; ++n) {
            const unsigned short* bp = bbase + n * 16 * LDSB_STRIDE;
            v8bf blo = *(const v8bf*)(bp);
            v8bf bhi = *(const v8bf*)(bp + 16);
            v16bf bfrag = __builtin_shufflevector(blo, bhi,
                0,1,2,3,4,5,6,7,8,9,10,11,12,13,14,15);
            // D = A x B + C  (neg_a, A, neg_b, B, c_mod, C, reuse_a, reuse_b)
            acc[n] = __builtin_amdgcn_wmma_f32_16x16x32_bf16(
                false, afrag, false, bfrag, (short)0, acc[n], false, false);
        }
    };

    // --- Double-buffered pipeline: issue k+1, wait own panel-k, barrier, --
    // --- compute on LDS, barrier (protects buffer reused by next issue). --
    issue_panel(0, 0);
    #pragma unroll
    for (int kk = 0; kk < 7; ++kk) {
        issue_panel((kk + 1) * 32, (kk + 1) & 1);
        WAIT_ASYNC(2);        // in-order completion => panel kk has landed
        __syncthreads();      // publish all waves' pieces of panel kk
        compute_step(kk * 32, kk & 1);
        __syncthreads();      // all reads of buf kk&1 done before overwrite
    }
    WAIT_ASYNC(0);
    __syncthreads();
    compute_step(7 * 32, 1);

    // --- Epilogue: sq = max(i2+t2-2g, 0); sim = exp(-sqrt(sq)/T) ----------
    // C/D layout: element r of acc lives at (M = r + 8*laneHi, N = laneLo).
    float i2v[8];
    #pragma unroll
    for (int r = 0; r < 8; ++r)
        i2v[r] = i2[rowBase + laneHi * 8 + r];

    float rs[8] = {};
    #pragma unroll
    for (int n = 0; n < 8; ++n) {
        const int col = colBase + n * 16 + laneLo;
        const float t2v = t2[col];
        #pragma unroll
        for (int r = 0; r < 8; ++r) {
            const int row = rowBase + laneHi * 8 + r;
            float sq   = fmaxf(i2v[r] + t2v - 2.0f * acc[n][r], 0.0f);
            float dist = sqrtf(sq);
            float sim  = __expf(-dist * INV_TEMP);
            rs[r] += sim;
            if (row == col) diag[row] = sim;   // exactly one writer per i
        }
    }

    // Horizontal reduce across the 16 N-lanes of each half-wave.
    // xor masks 1,2,4,8 never cross bit4, so the two halves stay separate.
    #pragma unroll
    for (int r = 0; r < 8; ++r) {
        float v = rs[r];
        v += __shfl_xor(v, 1, 32);
        v += __shfl_xor(v, 2, 32);
        v += __shfl_xor(v, 4, 32);
        v += __shfl_xor(v, 8, 32);
        if (laneLo == 0) {
            const int row = rowBase + laneHi * 8 + r;
            // Unique (blockIdx.y, row) slot -> deterministic, no atomics.
            partial[(size_t)blockIdx.y * B_SZ + row] = v;
        }
    }
}

// ---------------------------------------------------------------------------
// Kernel 3: fold 64 column-strip partials per row, compute loss, reduce.
// ---------------------------------------------------------------------------
__global__ void __launch_bounds__(1024)
finalize_kernel(const float* __restrict__ partial,  // [64][B_SZ]
                const float* __restrict__ diag,
                float* __restrict__ out) {
    __shared__ float smem[1024];
    const int tid = threadIdx.x;
    float s = 0.0f;
    for (int i = tid; i < B_SZ; i += 1024) {
        float num = diag[i];
        float den = 0.0f;
        #pragma unroll 4
        for (int p = 0; p < 64; ++p)
            den += partial[(size_t)p * B_SZ + i];
        den -= num;                               // exclude diagonal
        s += (den > 0.0f) ? -__logf(num / den) : 0.0f;
    }
    smem[tid] = s;
    __syncthreads();
    #pragma unroll
    for (int st = 512; st > 0; st >>= 1) {
        if (tid < st) smem[tid] += smem[tid + st];
        __syncthreads();
    }
    if (tid == 0) out[0] = smem[0] / (float)B_SZ;
}

// ---------------------------------------------------------------------------
extern "C" void kernel_launch(void* const* d_in, const int* in_sizes, int n_in,
                              void* d_out, int out_size, void* d_ws, size_t ws_size,
                              hipStream_t stream) {
    const float* text = (const float*)d_in[0];   // text_embeddings [B, D]
    const float* img  = (const float*)d_in[1];   // image_embeddings [B, D]

    // Workspace carve-up (~10.1 MB):
    //   textb : B*D bf16  (4 MB)
    //   imgb  : B*D bf16  (4 MB)
    //   t2,i2,diag : B floats each (96 KB)
    //   partial    : 64*B floats (2 MB)
    char* ws = (char*)d_ws;
    unsigned short* textb = (unsigned short*)ws;
    unsigned short* imgb  = (unsigned short*)(ws + (size_t)B_SZ * D_SZ * 2);
    float* t2      = (float*)(ws + (size_t)B_SZ * D_SZ * 4);
    float* i2      = t2 + B_SZ;
    float* diag    = i2 + B_SZ;
    float* partial = diag + B_SZ;

    prep_kernel<<<B_SZ, 256, 0, stream>>>(text, img, textb, imgb, t2, i2);

    dim3 grid(B_SZ / 128, B_SZ / 128);   // 64 x 64 blocks, 256 threads each
    gemm_sim_kernel<<<grid, 256, 0, stream>>>(imgb, textb, i2, t2, partial, diag);

    finalize_kernel<<<1, 1024, 0, stream>>>(partial, diag, (float*)d_out);
}
